// HebbFF_22789096472915
// MI455X (gfx1250) — matrix operand order repaired and checked by reading, compiled, not run on hardware
//
#include <hip/hip_runtime.h>
#include <hip/hip_bf16.h>
#include <math.h>

// CDNA5 / gfx1250: wave32, WMMA 16x16x32 bf16
typedef __attribute__((ext_vector_type(16))) __bf16 v16bf;
typedef __attribute__((ext_vector_type(8)))  float  v8f;

#define T_STEPS 128
#define BATCH   8
#define NH      512
#define DD      256

__device__ __forceinline__ float sigf(float v) { return 1.0f / (1.0f + expf(-v)); }

// One workgroup per batch sequence. 1024 threads = 32 waves.
// ws layout (floats): A_m [B][D][N] | A_a [B][M][N] | swzM bf16 | swzA bf16
__global__ __launch_bounds__(1024)
void hebbff_kernel(const float* __restrict__ x,       // [T,B,D]
                   const float* __restrict__ w_mult,  // [N,D]
                   const float* __restrict__ b_mult,  // [N]
                   const float* __restrict__ w_add,   // [N,N]
                   const float* __restrict__ b_add,   // [N]
                   const float* __restrict__ w_final, // [1,N]
                   const float* __restrict__ b_final, // [1]
                   const float* __restrict__ lmbda_mult,
                   const float* __restrict__ lmbda_add,
                   const float* __restrict__ eta_mult,
                   const float* __restrict__ eta_add,
                   float* __restrict__ out,           // [T,B,1]
                   float* __restrict__ ws)
{
    const int b    = blockIdx.x;
    const int tid  = threadIdx.x;
    const int lane = tid & 31;
    const int wave = tid >> 5;

    float*  Am   = ws;                                   // B*D*N
    float*  Aa   = ws + (size_t)BATCH * DD * NH;         // B*N*N
    __bf16* swzM = (__bf16*)(ws + (size_t)BATCH * DD * NH + (size_t)BATCH * NH * NH);
    __bf16* swzA = swzM + (size_t)NH * DD;

    float* AmB = Am + (size_t)b * DD * NH;               // [d][n]
    float* AaB = Aa + (size_t)b * NH * NH;               // [m][n]

    __shared__ __align__(32) float  xs[2][DD];
    __shared__ __align__(32) __bf16 xbf[DD];
    __shared__ __align__(32) float  hm[2][NH];
    __shared__ __align__(32) float  ha[2][NH];
    __shared__ __align__(32) __bf16 hbf[NH];
    __shared__ __align__(32) float  hdense[NH];
    __shared__ __align__(32) float  red[1024];

    // ---------------- init (every launch: deterministic) ----------------
    for (int i = tid; i < DD * NH; i += 1024) AmB[i] = 0.0f;
    for (int i = tid; i < NH * NH; i += 1024) AaB[i] = 0.0f;

    // Pre-swizzle weights into per-lane WMMA A tiles, contiguous k order (same as B reads).
    // element e of lane l: K = e + (l>=16?16:0)  (16x32 bf16 tile)
    for (int i = tid; i < NH * DD; i += 1024) {          // nkt = 8
        int e = i & 15, l = (i >> 4) & 31, tile = i >> 9;
        int kt = tile & 7, mt = tile >> 3;
        int row = mt * 16 + (l & 15);
        int k = kt * 32 + e + ((l >= 16) ? 16 : 0);
        swzM[i] = (__bf16)w_mult[(size_t)row * DD + k];
    }
    for (int i = tid; i < NH * NH; i += 1024) {          // nkt = 16
        int e = i & 15, l = (i >> 4) & 31, tile = i >> 9;
        int kt = tile & 15, mt = tile >> 4;
        int row = mt * 16 + (l & 15);
        int k = kt * 32 + e + ((l >= 16) ? 16 : 0);
        swzA[i] = (__bf16)w_add[(size_t)row * NH + k];
    }
    for (int i = tid; i < NH; i += 1024) { hm[0][i] = 0.f; hm[1][i] = 0.f; ha[0][i] = 0.f; ha[1][i] = 0.f; }
    for (int i = tid; i < DD; i += 1024) { xs[0][i] = 0.f; xs[1][i] = 0.f; }
    __syncthreads();

    const float lam_m = sigf(lmbda_mult[0]);
    const float lam_a = sigf(lmbda_add[0]);
    const float et_m  = eta_mult[0];
    const float et_a  = eta_add[0];

    const int n    = tid & 511;     // owned row
    const int half = tid >> 9;      // which half of the reduction dim

    for (int t = 0; t < T_STEPS; ++t) {
        const int cur = t & 1, prv = cur ^ 1;

        if (tid < DD) {
            float v = x[(size_t)(t * BATCH + b) * DD + tid];
            xs[cur][tid] = v;
            xbf[tid]     = (__bf16)v;
            if (t + 1 < T_STEPS)  // gfx1250 global_prefetch_b8 of next step's x
                __builtin_prefetch(&x[(size_t)((t + 1) * BATCH + b) * DD + tid], 0, 1);
        }
        __syncthreads();

        // ---- A1: dense term  w_mult @ x_t  via v_wmma_f32_16x16x32_bf16 ----
        {
            const int mt = wave;                                // 32 waves = 32 row tiles
            v8f c = {};
            const __bf16* wp = swzM + (size_t)mt * 8 * 512;
            for (int kt = 0; kt < 8; ++kt) {
                v16bf a  = *(const v16bf*)(wp + (size_t)kt * 512 + lane * 16);
                v16bf bb = *(const v16bf*)(&xbf[kt * 32 + 16 * (lane >> 4)]); // x in all 16 cols
                c = __builtin_amdgcn_wmma_f32_16x16x32_bf16(false, a, false, bb, (short)0, c, false, false);
            }
            if ((lane & 15) == 0) {
                for (int r = 0; r < 8; ++r)
                    hdense[mt * 16 + (lane >> 4) * 8 + r] = c[r];
            }
        }

        // ---- A2: streamed fused  A_m <- lam*A_m + eta*h_prev x_prev^T ;  acc = (w .* A_m) @ x ----
        {
            const float hprev = hm[prv][n];
            float acc = 0.0f;
            const int d0 = half * 128;
            for (int d = d0; d < d0 + 128; ++d) {
                float a = AmB[(size_t)d * NH + n];               // coalesced across lanes
                a = lam_m * a + et_m * hprev * xs[prv][d];
                AmB[(size_t)d * NH + n] = a;
                acc += w_mult[(size_t)n * DD + d] * a * xs[cur][d];
            }
            red[tid] = acc;
        }
        __syncthreads();
        if (tid < NH) {
            float h = tanhf(red[tid] + red[tid + 512] + hdense[tid] + b_mult[tid]);
            hm[cur][tid] = h;
            hbf[tid]     = (__bf16)h;
        }
        __syncthreads();

        // ---- B1: dense term  w_add @ h_m  via WMMA ----
        {
            const int mt = wave;
            v8f c = {};
            const __bf16* wp = swzA + (size_t)mt * 16 * 512;
            for (int kt = 0; kt < 16; ++kt) {
                v16bf a  = *(const v16bf*)(wp + (size_t)kt * 512 + lane * 16);
                v16bf bb = *(const v16bf*)(&hbf[kt * 32 + 16 * (lane >> 4)]);
                c = __builtin_amdgcn_wmma_f32_16x16x32_bf16(false, a, false, bb, (short)0, c, false, false);
            }
            if ((lane & 15) == 0) {
                for (int r = 0; r < 8; ++r)
                    hdense[mt * 16 + (lane >> 4) * 8 + r] = c[r];
            }
        }

        // ---- B2: streamed fused  A_a <- lam*A_a + eta*h_a_prev h_m_prev^T ;  acc = A_a @ h_m ----
        {
            const float haprev = ha[prv][n];
            float acc = 0.0f;
            const int m0 = half * 256;
            for (int m = m0; m < m0 + 256; ++m) {
                float a = AaB[(size_t)m * NH + n];               // coalesced across lanes
                a = lam_a * a + et_a * haprev * hm[prv][m];
                AaB[(size_t)m * NH + n] = a;
                acc += a * hm[cur][m];
            }
            red[tid] = acc;
        }
        __syncthreads();
        if (tid < NH)
            ha[cur][tid] = sigf(red[tid] + red[tid + 512] + hdense[tid] + b_add[tid]);
        __syncthreads();

        // ---- C: y = sigmoid(w_final . h_a + b_final) ----
        if (tid < NH) red[tid] = ha[cur][tid] * w_final[tid];
        __syncthreads();
        for (int s2 = 256; s2 > 0; s2 >>= 1) {
            if (tid < s2) red[tid] += red[tid + s2];
            __syncthreads();
        }
        if (tid == 0) out[t * BATCH + b] = sigf(red[0] + b_final[0]);
        __syncthreads();
    }
}

extern "C" void kernel_launch(void* const* d_in, const int* in_sizes, int n_in,
                              void* d_out, int out_size, void* d_ws, size_t ws_size,
                              hipStream_t stream) {
    const float* x          = (const float*)d_in[0];
    const float* w_mult     = (const float*)d_in[1];
    const float* b_mult     = (const float*)d_in[2];
    const float* w_add      = (const float*)d_in[3];
    const float* b_add      = (const float*)d_in[4];
    const float* w_final    = (const float*)d_in[5];
    const float* b_final    = (const float*)d_in[6];
    const float* lmbda_mult = (const float*)d_in[7];
    const float* lmbda_add  = (const float*)d_in[8];
    const float* eta_mult   = (const float*)d_in[9];
    const float* eta_add    = (const float*)d_in[10];
    // d_in[11], d_in[12]: Store/Batch flags (fixed 0/0 branch implemented)

    hipLaunchKernelGGL(hebbff_kernel, dim3(BATCH), dim3(1024), 0, stream,
                       x, w_mult, b_mult, w_add, b_add, w_final, b_final,
                       lmbda_mult, lmbda_add, eta_mult, eta_add,
                       (float*)d_out, (float*)d_ws);
}
